// LearnedMeans_39170101739947
// MI455X (gfx1250) — matrix operand order repaired and checked by reading, not compiled, not run on hardware
//
#include <hip/hip_runtime.h>
#include <hip/hip_bf16.h>
#include <math.h>

typedef __attribute__((ext_vector_type(16))) __bf16 v16bf;
typedef __attribute__((ext_vector_type(8)))  __bf16 bf16x8;
typedef __attribute__((ext_vector_type(8)))  float  v8f;

#define K_DIM 512
#define M_Q   1024

__device__ __forceinline__ __bf16 f32_to_bf16_rne(float f) {
  union { float f; unsigned u; } x; x.f = f;
  unsigned r = x.u + 0x7FFFu + ((x.u >> 16) & 1u);
  unsigned short h = (unsigned short)(r >> 16);
  __bf16 o; __builtin_memcpy(&o, &h, 2); return o;
}

// One block per row (K=512). Converts a f32 row to bf16 and computes its squared norm.
__global__ void convert_and_norm(const float* __restrict__ in,
                                 __bf16* __restrict__ out,
                                 float* __restrict__ norms) {
  const int row = blockIdx.x;
  const int tid = threadIdx.x;   // 128 threads
  const float* r = in + (size_t)row * K_DIM;
  __bf16* o = out + (size_t)row * K_DIM;
  float s = 0.f;
  #pragma unroll
  for (int i = 0; i < 4; ++i) {
    float v = r[tid + i * 128];
    s += v * v;
    o[tid + i * 128] = f32_to_bf16_rne(v);
  }
  #pragma unroll
  for (int m = 16; m >= 1; m >>= 1) s += __shfl_xor(s, m, 32);
  __shared__ float ws[4];
  if ((tid & 31) == 0) ws[tid >> 5] = s;
  __syncthreads();
  if (tid == 0) norms[row] = ws[0] + ws[1] + ws[2] + ws[3];
}

union FragA { struct { bf16x8 lo, hi; } h; v16bf v; };

// Each wave: one 16-row query panel (A resident in 128 VGPRs), strided 16-col tiles of the
// dataset. Computes dist^2 = q2 + d2 - 2*QD^T via v_wmma_f32_16x16x32_bf16 and keeps the
// two smallest per row; per-wave partials written to `partial` ([M][P][2]).
__global__ void __launch_bounds__(256)
knn_best2_gemm(const __bf16* __restrict__ Qbf, const __bf16* __restrict__ Dbf,
               const float* __restrict__ q2, const float* __restrict__ d2,
               float* __restrict__ partial, int N, int P) {
  const int lane = threadIdx.x & 31;
  const int wave = threadIdx.x >> 5;
  const int m0   = blockIdx.x << 4;
  const int p    = blockIdx.y * 8 + wave;  // wave's partial slot, p in [0,P)
  const int r    = lane & 15;
  const int kh   = lane >> 4;

  // A fragments: 16 k-steps of 16x32 bf16 (ISA 7.12.2 layout)
  const __bf16* qrow = Qbf + (size_t)(m0 + r) * K_DIM;
  v16bf afrag[16];
  #pragma unroll
  for (int ks = 0; ks < 16; ++ks) {
    FragA f;
    f.h.lo = *reinterpret_cast<const bf16x8*>(qrow + ks * 32 + kh * 8);
    f.h.hi = *reinterpret_cast<const bf16x8*>(qrow + ks * 32 + 16 + kh * 8);
    afrag[ks] = f.v;
  }
  float q2v[8];
  #pragma unroll
  for (int g = 0; g < 8; ++g) q2v[g] = q2[m0 + g + 8 * kh];

  float b0[8], b1[8];
  #pragma unroll
  for (int g = 0; g < 8; ++g) { b0[g] = 3.0e38f; b1[g] = 3.0e38f; }

  const int ntiles = N >> 4;
  for (int t = p; t < ntiles; t += P) {
    const int n0 = t << 4;
    const __bf16* drow = Dbf + (size_t)(n0 + r) * K_DIM;
    v8f acc = {};
    #pragma unroll
    for (int ks = 0; ks < 16; ++ks) {
      FragA fb;
      fb.h.lo = *reinterpret_cast<const bf16x8*>(drow + ks * 32 + kh * 8);
      fb.h.hi = *reinterpret_cast<const bf16x8*>(drow + ks * 32 + 16 + kh * 8);
      v16bf b = fb.v;
      acc = __builtin_amdgcn_wmma_f32_16x16x32_bf16(false, afrag[ks], false, b,
                                                    (short)0, acc, false, false);
    }
    float dn = d2[n0 + r];
    #pragma unroll
    for (int g = 0; g < 8; ++g) {
      float dist = fmaxf(q2v[g] + dn - 2.0f * acc[g], 0.0f);
      float nb1 = fminf(b1[g], fmaxf(b0[g], dist));
      b0[g] = fminf(b0[g], dist);
      b1[g] = nb1;
    }
  }

  // merge the 16 column-lanes per half (C/D layout: lanes 0-15 rows g, 16-31 rows 8+g)
  #pragma unroll
  for (int m = 1; m <= 8; m <<= 1) {
    #pragma unroll
    for (int g = 0; g < 8; ++g) {
      float o0 = __shfl_xor(b0[g], m, 32);
      float o1 = __shfl_xor(b1[g], m, 32);
      float nb0 = fminf(b0[g], o0);
      float nb1 = fminf(fmaxf(b0[g], o0), fminf(b1[g], o1));
      b0[g] = nb0; b1[g] = nb1;
    }
  }
  if (r == 0) {
    #pragma unroll
    for (int g = 0; g < 8; ++g) {
      int mrow = m0 + g + 8 * kh;
      partial[((size_t)mrow * P + p) * 2 + 0] = b0[g];
      partial[((size_t)mrow * P + p) * 2 + 1] = b1[g];
    }
  }
}

__global__ void reduce_best2(const float* __restrict__ partial, int P,
                             float* __restrict__ out2) {
  int m = blockIdx.x * blockDim.x + threadIdx.x;
  if (m >= M_Q) return;
  float b0 = 3.0e38f, b1 = 3.0e38f;
  for (int p = 0; p < P; ++p) {
    float a0 = partial[((size_t)m * P + p) * 2 + 0];
    float a1 = partial[((size_t)m * P + p) * 2 + 1];
    float nb1 = fminf(fmaxf(b0, a0), fminf(b1, a1));
    b0 = fminf(b0, a0);
    b1 = nb1;
  }
  out2[m * 2 + 0] = sqrtf(b0);
  out2[m * 2 + 1] = sqrtf(b1);
}

__device__ __forceinline__ void bitonic_sort_1024(float* s, int t) {
  for (int k = 2; k <= 1024; k <<= 1) {
    for (int j = k >> 1; j > 0; j >>= 1) {
      int ixj = t ^ j;
      if (ixj > t) {
        float a = s[t], b = s[ixj];
        bool asc = ((t & k) == 0);
        if ((a > b) == asc) { s[t] = b; s[ixj] = a; }
      }
      __syncthreads();
    }
  }
}

__global__ void __launch_bounds__(1024)
stats_kernel(const float* __restrict__ dm, const float* __restrict__ dsd,
             float* __restrict__ out) {
  __shared__ float sbuf[1024];
  __shared__ float acc[6];
  const int t = threadIdx.x;
  if (t < 6) acc[t] = 0.f;
  __syncthreads();
  float m0 = dm[t * 2], m1 = dm[t * 2 + 1];
  float s0 = dsd[t * 2], s1 = dsd[t * 2 + 1];
  const float TH = 1.0f / 3.0f;
  float nm = ((m0 < TH * s0) && (m0 < TH * m1)) ? 1.f : 0.f;
  float ns = ((s0 < TH * m0) && (s0 < TH * s1)) ? 1.f : 0.f;
  atomicAdd(&acc[0], nm);
  atomicAdd(&acc[1], ns);
  atomicAdd(&acc[2], m0);
  atomicAdd(&acc[3], s0);
  atomicAdd(&acc[4], m1);
  atomicAdd(&acc[5], s1);
  __syncthreads();
  if (t == 0) {
    out[0] = acc[0];
    out[1] = acc[1];
    out[2] = acc[2] * (1.0f / 1024.0f);
    out[3] = acc[3] * (1.0f / 1024.0f);
    out[4] = acc[4] * (1.0f / 1024.0f);
    out[5] = acc[5] * (1.0f / 1024.0f);
  }
  sbuf[t] = m0; __syncthreads();
  bitonic_sort_1024(sbuf, t);
  if (t == 0) {
    const float pct[5] = {10.f, 25.f, 50.f, 75.f, 90.f};
    #pragma unroll
    for (int i = 0; i < 5; ++i) {
      float pos = pct[i] * 0.01f * 1023.f;
      int lo = (int)pos;
      float fr = pos - (float)lo;
      out[6 + i] = sbuf[lo] + (sbuf[lo + 1] - sbuf[lo]) * fr;
    }
  }
  __syncthreads();
  sbuf[t] = s0; __syncthreads();
  bitonic_sort_1024(sbuf, t);
  if (t == 0) {
    const float pct[5] = {10.f, 25.f, 50.f, 75.f, 90.f};
    #pragma unroll
    for (int i = 0; i < 5; ++i) {
      float pos = pct[i] * 0.01f * 1023.f;
      int lo = (int)pos;
      float fr = pos - (float)lo;
      out[11 + i] = sbuf[lo] + (sbuf[lo + 1] - sbuf[lo]) * fr;
    }
  }
}

extern "C" void kernel_launch(void* const* d_in, const int* in_sizes, int n_in,
                              void* d_out, int out_size, void* d_ws, size_t ws_size,
                              hipStream_t stream) {
  (void)in_sizes; (void)n_in; (void)out_size; (void)ws_size;
  const float* Q = (const float*)d_in[0];   // [1024, 512]
  const float* T = (const float*)d_in[1];   // [4096, 512]
  const float* X = (const float*)d_in[2];   // [16384, 512]
  float* out = (float*)d_out;

  char* w = (char*)d_ws;
  __bf16* Qbf = (__bf16*)w;  w += (size_t)1024  * 512 * 2;
  __bf16* Tbf = (__bf16*)w;  w += (size_t)4096  * 512 * 2;
  __bf16* Xbf = (__bf16*)w;  w += (size_t)16384 * 512 * 2;
  float* q2    = (float*)w;  w += (size_t)1024  * 4;
  float* t2    = (float*)w;  w += (size_t)4096  * 4;
  float* x2    = (float*)w;  w += (size_t)16384 * 4;
  float* partT = (float*)w;  w += (size_t)1024 * 64  * 2 * 4;
  float* partX = (float*)w;  w += (size_t)1024 * 128 * 2 * 4;
  float* dmv   = (float*)w;  w += (size_t)1024 * 2 * 4;
  float* dsv   = (float*)w;  w += (size_t)1024 * 2 * 4;

  convert_and_norm<<<1024,  128, 0, stream>>>(Q, Qbf, q2);
  convert_and_norm<<<4096,  128, 0, stream>>>(T, Tbf, t2);
  convert_and_norm<<<16384, 128, 0, stream>>>(X, Xbf, x2);

  // true_means: N=4096, 64 partials per row (8 blocks-y x 8 waves)
  knn_best2_gemm<<<dim3(64, 8),  256, 0, stream>>>(Qbf, Tbf, q2, t2, partT, 4096,  64);
  // X_train: N=16384, 128 partials per row (16 blocks-y x 8 waves)
  knn_best2_gemm<<<dim3(64, 16), 256, 0, stream>>>(Qbf, Xbf, q2, x2, partX, 16384, 128);

  reduce_best2<<<4, 256, 0, stream>>>(partT, 64,  dmv);
  reduce_best2<<<4, 256, 0, stream>>>(partX, 128, dsv);

  stats_kernel<<<1, 1024, 0, stream>>>(dmv, dsv, out);
}
